// Qwen3MHAttention_64312840290532
// MI455X (gfx1250) — matrix operand order repaired and checked by reading, not compiled, not run on hardware
//
#include <hip/hip_runtime.h>
#include <hip/hip_bf16.h>

// ---------------------------------------------------------------------------
// Qwen3 attention block on gfx1250 (MI455X), wave32 + v_wmma_f32_16x16x32_bf16
// ---------------------------------------------------------------------------

#define DEV static __device__ __forceinline__

typedef __attribute__((ext_vector_type(16))) __bf16 v16bf;
typedef __attribute__((ext_vector_type(8)))  float  v8f;
typedef __attribute__((ext_vector_type(4)))  unsigned int u32x4;

union BfFrag {
    v16bf v;
    unsigned short us[16];
    u32x4 q[2];
};

// constants of the problem (fixed by the reference)
constexpr int Bc  = 2;
constexpr int Tc  = 2048;
constexpr int Dc  = 2048;
constexpr int Hc  = 16;
constexpr int HKVc = 8;
constexpr int DHc = 128;

DEV unsigned short f2bf(float f) {
    union { float f; unsigned u; } v; v.f = f;
    unsigned u = v.u;
    u += 0x7FFFu + ((u >> 16) & 1u);   // round-to-nearest-even
    return (unsigned short)(u >> 16);
}
DEV float bf2f(unsigned short u) {
    union { float f; unsigned u; } v; v.u = ((unsigned)u) << 16;
    return v.f;
}

DEV v8f wmma_bf16(const v16bf& a, const v16bf& b, const v8f& c) {
    return __builtin_amdgcn_wmma_f32_16x16x32_bf16(false, a, false, b,
                                                   (short)0, c, false, false);
}

// ---------------------------------------------------------------------------
// 1) fp32 -> bf16 conversion
// ---------------------------------------------------------------------------
__global__ __launch_bounds__(256) void cvt_f32_bf16(const float* __restrict__ in,
                                                    unsigned short* __restrict__ out,
                                                    int n) {
    int i = blockIdx.x * blockDim.x + threadIdx.x;
    int stride = gridDim.x * blockDim.x;
    for (; i < n; i += stride) out[i] = f2bf(in[i]);
}

// ---------------------------------------------------------------------------
// 2) GEMM: C[M,N] = A[M,K] (bf16, row-major) @ W[N,K]^T (bf16, row-major)
//    block = 256 thr = 8 waves (4x2) -> 128(M) x 64(N) tile
//    each wave: 32x32 output = 2x2 WMMA register blocking
//      -> 8 b128 loads feed 4 WMMAs (2 loads/WMMA, 2x intensity vs 1x1)
//    mode 0: out fp32 [M,N]
//    mode 2: out bf16 reshaped to [B, Hx, T, DH]  (for q/k/v)
//    Both fragment kinds are contiguous 16B global loads -> no LDS needed.
// ---------------------------------------------------------------------------
DEV void store_tile(void* Out, int N, int mode, int Hx,
                    int row, int col, const v8f& acc, int lo, int hi) {
    if (mode == 0) {
        float* o = (float*)Out;
        #pragma unroll
        for (int r = 0; r < 8; ++r)
            o[(size_t)(row + r + 8 * hi) * N + col + lo] = acc[r];
    } else { // mode 2: n -> (head, dh), m -> (b, t); out[b][head][t][dh] bf16
        unsigned short* o = (unsigned short*)Out;
        #pragma unroll
        for (int r = 0; r < 8; ++r) {
            int m = row + r + 8 * hi;
            int bb = m >> 11;            // / T
            int t  = m & (Tc - 1);
            int n  = col + lo;
            int head = n >> 7;           // / DH
            int dh   = n & (DHc - 1);
            o[(((size_t)bb * Hx + head) * Tc + t) * DHc + dh] = f2bf(acc[r]);
        }
    }
}

__global__ __launch_bounds__(256) void gemm_bf16(const unsigned short* __restrict__ A,
                                                 const unsigned short* __restrict__ W,
                                                 void* __restrict__ Out,
                                                 int M, int N, int K,
                                                 int mode, int Hx) {
    const int lane = threadIdx.x & 31;
    const int w    = threadIdx.x >> 5;
    const int wm   = w >> 1, wn = w & 1;
    const int row0 = blockIdx.y * 128 + wm * 32;
    const int col0 = blockIdx.x * 64 + wn * 32;
    const int lo = lane & 15, hi = lane >> 4;

    const unsigned short* ap0 = A + (size_t)(row0 + lo) * K;            // lane = M row
    const unsigned short* ap1 = ap0 + (size_t)16 * K;
    const unsigned short* wp0 = W + (size_t)(col0 + lo) * K + 16 * hi;  // lane = N row
    const unsigned short* wp1 = wp0 + (size_t)16 * K;

    v8f acc00 = {}, acc01 = {}, acc10 = {}, acc11 = {};
    for (int kc = 0; kc < K; kc += 32) {
        __builtin_prefetch(ap0 + kc + 256, 0, 1);
        __builtin_prefetch(ap1 + kc + 256, 0, 1);
        __builtin_prefetch(wp0 + kc + 256, 0, 1);
        __builtin_prefetch(wp1 + kc + 256, 0, 1);
        BfFrag a0, a1, b0, b1;
        // A frag: u[j] -> k = kc+16*hi+j  (lane = m)
        a0.q[0] = *(const u32x4*)(ap0 + kc + 16 * hi);
        a0.q[1] = *(const u32x4*)(ap0 + kc + 16 * hi + 8);
        a1.q[0] = *(const u32x4*)(ap1 + kc + 16 * hi);
        a1.q[1] = *(const u32x4*)(ap1 + kc + 16 * hi + 8);
        // B frag: u[j] -> k = kc+16*hi+j  (lane = n)
        b0.q[0] = *(const u32x4*)(wp0 + kc);
        b0.q[1] = *(const u32x4*)(wp0 + kc + 8);
        b1.q[0] = *(const u32x4*)(wp1 + kc);
        b1.q[1] = *(const u32x4*)(wp1 + kc + 8);
        acc00 = wmma_bf16(a0.v, b0.v, acc00);
        acc01 = wmma_bf16(a0.v, b1.v, acc01);
        acc10 = wmma_bf16(a1.v, b0.v, acc10);
        acc11 = wmma_bf16(a1.v, b1.v, acc11);
    }

    store_tile(Out, N, mode, Hx, row0,      col0,      acc00, lo, hi);
    store_tile(Out, N, mode, Hx, row0,      col0 + 16, acc01, lo, hi);
    store_tile(Out, N, mode, Hx, row0 + 16, col0,      acc10, lo, hi);
    store_tile(Out, N, mode, Hx, row0 + 16, col0 + 16, acc11, lo, hi);
}

// ---------------------------------------------------------------------------
// 3) RMSNorm + RoPE, in place on q [B,H,T,DH] and k [B,HKV,T,DH] (bf16)
//    one wave per row; lane owns d = {l, l+32, l+64, l+96} -> rope pairs local
// ---------------------------------------------------------------------------
__global__ __launch_bounds__(256) void rmsnorm_rope(unsigned short* __restrict__ qb,
                                                    unsigned short* __restrict__ kb,
                                                    const float* __restrict__ qw,
                                                    const float* __restrict__ kw) {
    const int ROWS_Q = Bc * Hc * Tc;       // 65536
    const int ROWS_K = Bc * HKVc * Tc;     // 32768
    int wave = blockIdx.x * 8 + (threadIdx.x >> 5);
    if (wave >= ROWS_Q + ROWS_K) return;
    int lane = threadIdx.x & 31;

    unsigned short* base;
    const float* wgt;
    int t;
    if (wave < ROWS_Q) { base = qb + (size_t)wave * DHc; wgt = qw; t = wave & (Tc - 1); }
    else { int r = wave - ROWS_Q; base = kb + (size_t)r * DHc; wgt = kw; t = r & (Tc - 1); }

    float x0 = bf2f(base[lane]);
    float x1 = bf2f(base[lane + 32]);
    float x2 = bf2f(base[lane + 64]);
    float x3 = bf2f(base[lane + 96]);

    float ss = x0 * x0 + x1 * x1 + x2 * x2 + x3 * x3;
    ss += __shfl_xor(ss, 1, 32);
    ss += __shfl_xor(ss, 2, 32);
    ss += __shfl_xor(ss, 4, 32);
    ss += __shfl_xor(ss, 8, 32);
    ss += __shfl_xor(ss, 16, 32);
    float rms = rsqrtf(ss * (1.0f / 128.0f) + 1e-6f);

    float n0 = x0 * rms * wgt[lane];
    float n1 = x1 * rms * wgt[lane + 32];
    float n2 = x2 * rms * wgt[lane + 64];
    float n3 = x3 * rms * wgt[lane + 96];

    // inv_freq[i] = 1e6^(-i/64) = exp2(-i * log2(1e6)/64)
    const float l2t_64 = 19.9315685693241740f / 64.0f;
    float a0 = (float)t * exp2f(-(float)lane * l2t_64);
    float a1 = (float)t * exp2f(-(float)(lane + 32) * l2t_64);
    float c0 = cosf(a0), s0 = sinf(a0);
    float c1 = cosf(a1), s1 = sinf(a1);

    base[lane]      = f2bf(n0 * c0 - n2 * s0);   // d < 64 : x*cos - x[d+64]*sin
    base[lane + 64] = f2bf(n2 * c0 + n0 * s0);   // d >=64 : x*cos + x[d-64]*sin
    base[lane + 32] = f2bf(n1 * c1 - n3 * s1);
    base[lane + 96] = f2bf(n3 * c1 + n1 * s1);
}

// ---------------------------------------------------------------------------
// 4) Flash attention (causal, GQA G=2), online softmax, all-WMMA
//    grid = (T/128, B*H); block = 256 = 8 waves; wave owns 16 q rows.
// ---------------------------------------------------------------------------
__global__ __launch_bounds__(256) void flash_attn(const unsigned short* __restrict__ q,
                                                  const unsigned short* __restrict__ k,
                                                  const unsigned short* __restrict__ v,
                                                  unsigned short* __restrict__ ctx) {
    __shared__ __align__(16) unsigned short vT[128][40];   // V tile transposed [d][key]
    __shared__ __align__(16) unsigned short Pl[8][16][40]; // per-wave P staging

    const int lane = threadIdx.x & 31;
    const int w    = threadIdx.x >> 5;
    const int lo = lane & 15, hi = lane >> 4;
    const int bh = blockIdx.y;
    const int b   = bh >> 4;          // / H
    const int h   = bh & (Hc - 1);
    const int kvh = h >> 1;           // GQA group of 2
    const int qbase = blockIdx.x * 128 + w * 16;

    const unsigned short* qhead = q + ((size_t)b * Hc + h) * Tc * DHc;
    const unsigned short* khead = k + ((size_t)b * HKVc + kvh) * Tc * DHc;
    const unsigned short* vhead = v + ((size_t)b * HKVc + kvh) * Tc * DHc;

    // Q fragments for the whole DH (4 chunks of 32), direct contiguous loads
    BfFrag aq[4];
    {
        const unsigned short* qp = qhead + (size_t)(qbase + lo) * DHc;
        #pragma unroll
        for (int c = 0; c < 4; ++c) {
            aq[c].q[0] = *(const u32x4*)(qp + c * 32 + 16 * hi);
            aq[c].q[1] = *(const u32x4*)(qp + c * 32 + 16 * hi + 8);
        }
    }

    v8f o[8];
    #pragma unroll
    for (int c = 0; c < 8; ++c) o[c] = (v8f){};
    float mrow[8], lrow[8];
    #pragma unroll
    for (int r = 0; r < 8; ++r) { mrow[r] = -1e30f; lrow[r] = 0.0f; }

    const int kmax = (blockIdx.x + 1) * 128;
    for (int kt = 0; kt < kmax; kt += 32) {
        __syncthreads();
        { // cooperative stage: transpose 32x128 V tile into LDS
            int key = threadIdx.x >> 3;
            int dg  = (threadIdx.x & 7) * 16;
            const unsigned short* vp = vhead + (size_t)(kt + key) * DHc + dg;
            union { u32x4 q[2]; unsigned short us[16]; } tb;
            tb.q[0] = *(const u32x4*)(vp);
            tb.q[1] = *(const u32x4*)(vp + 8);
            #pragma unroll
            for (int j = 0; j < 16; ++j) vT[dg + j][key] = tb.us[j];
        }
        __syncthreads();
        if (kt > qbase + 15) continue;   // fully masked for this wave

        // ---- S = Q K^T over 32 keys (two 16-key C tiles) ----
        v8f s0 = {}, s1 = {};
        #pragma unroll
        for (int c = 0; c < 4; ++c) {
            BfFrag b0, b1;  // lane = key; 16 consecutive d from K row
            const unsigned short* kp0 = khead + (size_t)(kt + lo) * DHc + c * 32 + 16 * hi;
            const unsigned short* kp1 = khead + (size_t)(kt + 16 + lo) * DHc + c * 32 + 16 * hi;
            b0.q[0] = *(const u32x4*)(kp0);  b0.q[1] = *(const u32x4*)(kp0 + 8);
            b1.q[0] = *(const u32x4*)(kp1);  b1.q[1] = *(const u32x4*)(kp1 + 8);
            s0 = wmma_bf16(aq[c].v, b0.v, s0);
            s1 = wmma_bf16(aq[c].v, b1.v, s1);
        }

        // ---- masked online softmax (rows map to r + 8*hi for S, P and O) ----
        const float sc = 0.08838834764831845f; // 1/sqrt(128)
        float p0[8], p1[8];
        #pragma unroll
        for (int r = 0; r < 8; ++r) {
            int qi = qbase + r + 8 * hi;
            float e0 = (kt + lo      <= qi) ? s0[r] * sc : -1e30f;
            float e1 = (kt + 16 + lo <= qi) ? s1[r] * sc : -1e30f;
            float rv = fmaxf(e0, e1);
            rv = fmaxf(rv, __shfl_xor(rv, 1, 16));
            rv = fmaxf(rv, __shfl_xor(rv, 2, 16));
            rv = fmaxf(rv, __shfl_xor(rv, 4, 16));
            rv = fmaxf(rv, __shfl_xor(rv, 8, 16));
            float mn = fmaxf(mrow[r], rv);
            float scale = __expf(mrow[r] - mn);
            mrow[r] = mn;
            e0 = __expf(e0 - mn);
            e1 = __expf(e1 - mn);
            float ts = e0 + e1;
            ts += __shfl_xor(ts, 1, 16);
            ts += __shfl_xor(ts, 2, 16);
            ts += __shfl_xor(ts, 4, 16);
            ts += __shfl_xor(ts, 8, 16);
            lrow[r] = lrow[r] * scale + ts;
            p0[r] = e0; p1[r] = e1;
            #pragma unroll
            for (int c = 0; c < 8; ++c) o[c][r] *= scale;
        }

        // ---- relayout P (C-frag, lane=col) -> A-frag (lane=row) via wave LDS ----
        #pragma unroll
        for (int r = 0; r < 8; ++r) {
            Pl[w][r + 8 * hi][lo]      = f2bf(p0[r]);
            Pl[w][r + 8 * hi][16 + lo] = f2bf(p1[r]);
        }
        asm volatile("s_wait_dscnt 0" ::: "memory");   // wave-lockstep LDS sync
        BfFrag pa;
        pa.q[0] = *(const u32x4*)(&Pl[w][lo][16 * hi]);
        pa.q[1] = *(const u32x4*)(&Pl[w][lo][16 * hi + 8]);

        // ---- O += P @ V  (8 chunks of 16 output dims) ----
        #pragma unroll
        for (int c = 0; c < 8; ++c) {
            BfFrag bv;  // lane = d; 16 consecutive keys from vT row
            bv.q[0] = *(const u32x4*)(&vT[c * 16 + lo][16 * hi]);
            bv.q[1] = *(const u32x4*)(&vT[c * 16 + lo][16 * hi + 8]);
            o[c] = wmma_bf16(pa.v, bv.v, o[c]);
        }
    }

    // epilogue: normalize and write ctx [B, T, H*DH] as bf16
    unsigned short* op = ctx + ((size_t)b * Tc + qbase) * (Hc * DHc) + h * DHc;
    #pragma unroll
    for (int c = 0; c < 8; ++c) {
        #pragma unroll
        for (int r = 0; r < 8; ++r) {
            float val = o[c][r] / lrow[r];
            op[(size_t)(r + 8 * hi) * (Hc * DHc) + c * 16 + lo] = f2bf(val);
        }
    }
}

// ---------------------------------------------------------------------------
// host: orchestrate the pipeline on `stream`
// ---------------------------------------------------------------------------
extern "C" void kernel_launch(void* const* d_in, const int* in_sizes, int n_in,
                              void* d_out, int out_size, void* d_ws, size_t ws_size,
                              hipStream_t stream) {
    const float* x  = (const float*)d_in[0];
    const float* wq = (const float*)d_in[1];
    const float* wk = (const float*)d_in[2];
    const float* wv = (const float*)d_in[3];
    const float* wo = (const float*)d_in[4];
    const float* qw = (const float*)d_in[5];
    const float* kw = (const float*)d_in[6];

    size_t off = 0;
    auto carve = [&](size_t bytes) {
        void* p = (char*)d_ws + off;
        off += (bytes + 255) & ~(size_t)255;
        return p;
    };
    const size_t M  = (size_t)Bc * Tc;          // 4096
    unsigned short* xbf  = (unsigned short*)carve(M * Dc * 2);
    unsigned short* wqbf = (unsigned short*)carve((size_t)Hc * DHc * Dc * 2);
    unsigned short* wkbf = (unsigned short*)carve((size_t)HKVc * DHc * Dc * 2);
    unsigned short* wvbf = (unsigned short*)carve((size_t)HKVc * DHc * Dc * 2);
    unsigned short* wobf = (unsigned short*)carve((size_t)Dc * Hc * DHc * 2);
    unsigned short* qbuf = (unsigned short*)carve(M * Hc * DHc * 2);
    unsigned short* kbuf = (unsigned short*)carve(M * HKVc * DHc * 2);
    unsigned short* vbuf = (unsigned short*)carve(M * HKVc * DHc * 2);
    unsigned short* cbuf = (unsigned short*)carve(M * Hc * DHc * 2);

    // 1) convert activations + weights to bf16
    cvt_f32_bf16<<<4096, 256, 0, stream>>>(x,  xbf,  (int)(M * Dc));
    cvt_f32_bf16<<<2048, 256, 0, stream>>>(wq, wqbf, Hc * DHc * Dc);
    cvt_f32_bf16<<<2048, 256, 0, stream>>>(wk, wkbf, HKVc * DHc * Dc);
    cvt_f32_bf16<<<2048, 256, 0, stream>>>(wv, wvbf, HKVc * DHc * Dc);
    cvt_f32_bf16<<<2048, 256, 0, stream>>>(wo, wobf, Dc * Hc * DHc);

    // 2) QKV projections (write reshaped [B, Hx, T, DH] bf16)
    gemm_bf16<<<dim3((Hc * DHc) / 64, M / 128), 256, 0, stream>>>(
        xbf, wqbf, qbuf, (int)M, Hc * DHc, Dc, 2, Hc);
    gemm_bf16<<<dim3((HKVc * DHc) / 64, M / 128), 256, 0, stream>>>(
        xbf, wkbf, kbuf, (int)M, HKVc * DHc, Dc, 2, HKVc);
    gemm_bf16<<<dim3((HKVc * DHc) / 64, M / 128), 256, 0, stream>>>(
        xbf, wvbf, vbuf, (int)M, HKVc * DHc, Dc, 2, HKVc);

    // 3) RMSNorm + RoPE on q and k (in place)
    {
        int waves = Bc * (Hc + HKVc) * Tc;     // 98304
        rmsnorm_rope<<<(waves + 7) / 8, 256, 0, stream>>>(qbuf, kbuf, qw, kw);
    }

    // 4) causal GQA flash attention -> ctx [B, T, H*DH] bf16
    flash_attn<<<dim3(Tc / 128, Bc * Hc), 256, 0, stream>>>(qbuf, kbuf, vbuf, cbuf);

    // 5) output projection -> d_out fp32 [B*T, D]
    gemm_bf16<<<dim3(Dc / 64, M / 128), 256, 0, stream>>>(
        cbuf, wobf, d_out, (int)M, Dc, Hc * DHc, 0, 0);
}
